// MultiHeadAttentionPure_5488968204519
// MI455X (gfx1250) — matrix operand order repaired and checked by reading, not compiled, not run on hardware
//
#include <hip/hip_runtime.h>
#include <hip/hip_bf16.h>

typedef __attribute__((ext_vector_type(16))) __bf16 bf16x16;
typedef __attribute__((ext_vector_type(8)))  float  floatx8;

#define SEQ  1024
#define FDIM 1024
#define HD   16
#define NB   2
#define NCH  64   // FDIM / HD

__device__ __forceinline__ floatx8 wmma_bf16(bf16x16 a, bf16x16 b, floatx8 c) {
  return __builtin_amdgcn_wmma_f32_16x16x32_bf16(false, a, false, b, (short)0, c,
                                                 false, false);
}

// Convert 8 contiguous floats (16B-aligned) to bf16 into d[o..o+7]
__device__ __forceinline__ void cvt8(const float* __restrict__ p, bf16x16& d, int o) {
  const float4 f0 = *(const float4*)(p);
  const float4 f1 = *(const float4*)(p + 4);
  d[o + 0] = (__bf16)f0.x; d[o + 1] = (__bf16)f0.y;
  d[o + 2] = (__bf16)f0.z; d[o + 3] = (__bf16)f0.w;
  d[o + 4] = (__bf16)f1.x; d[o + 5] = (__bf16)f1.y;
  d[o + 6] = (__bf16)f1.z; d[o + 7] = (__bf16)f1.w;
}

// ---- 16-lane row reductions on the VALU via DPP row_ror (stays in DPP row,
// which is exactly the lane group used by the WMMA C/D layout on wave32) ----
#define DPP_STEP_MAX(x, ctrl)                                                  \
  {                                                                            \
    int _yi = __builtin_amdgcn_mov_dpp(__builtin_bit_cast(int, x), ctrl, 0xF,  \
                                       0xF, false);                            \
    x = fmaxf(x, __builtin_bit_cast(float, _yi));                              \
  }
#define DPP_STEP_ADD(x, ctrl)                                                  \
  {                                                                            \
    int _yi = __builtin_amdgcn_mov_dpp(__builtin_bit_cast(int, x), ctrl, 0xF,  \
                                       0xF, false);                            \
    x = x + __builtin_bit_cast(float, _yi);                                    \
  }

__device__ __forceinline__ float row16_max(float x) {
  DPP_STEP_MAX(x, 0x121)  // row_ror:1
  DPP_STEP_MAX(x, 0x122)  // row_ror:2
  DPP_STEP_MAX(x, 0x124)  // row_ror:4
  DPP_STEP_MAX(x, 0x128)  // row_ror:8
  return x;
}
__device__ __forceinline__ float row16_sum(float x) {
  DPP_STEP_ADD(x, 0x121)
  DPP_STEP_ADD(x, 0x122)
  DPP_STEP_ADD(x, 0x124)
  DPP_STEP_ADD(x, 0x128)
  return x;
}

union BF16x16U { bf16x16 v; uint4 q[2]; };

// ---------------------------------------------------------------------------
// Kernel 1: fused causal attention per (head-chunk j, batch b, 16-query tile)
// One wave32 per tile. Head dim = 16 (padded to WMMA K=32 on the QK^T side).
// ---------------------------------------------------------------------------
__global__ __launch_bounds__(32) void attn_fused(
    const float* __restrict__ q, const float* __restrict__ k,
    const float* __restrict__ v, const int* __restrict__ amask,
    float* __restrict__ x)
{
  __shared__ __bf16 lp[16 * 40];          // P staging, stride 40 halves (pad)

  const int lane = threadIdx.x & 31;
  const int g    = lane >> 4;             // lane group (0/1)
  const int ln   = lane & 15;

  const int qt = blockIdx.x;              // query tile 0..63
  const int b  = blockIdx.y;              // batch 0..1
  const int j  = blockIdx.z;              // head chunk 0..63
  const int q0 = qt * 16;
  const int mask = amask[0];

  const size_t base = ((size_t)b * SEQ) * FDIM + (size_t)j * HD;
  const float* __restrict__ qb = q + base;
  const float* __restrict__ kb = k + base;
  const float* __restrict__ vb = v + base;

  // ---- Q tile -> A layout (16x32 bf16, K=16..31 zero-padded) ----
  // A layout: lane (m = ln); elems 0..7 hold K = 8*g + 0..7; elems 8..15 zero
  bf16x16 Aq;
  cvt8(qb + (size_t)(q0 + ln) * FDIM + 8 * g, Aq, 0);
  #pragma unroll
  for (int e = 8; e < 16; ++e) Aq[e] = (__bf16)0.0f;

  floatx8 O;
  float Mrun[8], Lpart[8];                // Lpart is LANE-LOCAL partial sum
  #pragma unroll
  for (int r = 0; r < 8; ++r) { O[r] = 0.0f; Mrun[r] = -3.0e38f; Lpart[r] = 0.0f; }

  floatx8 zero8;
  #pragma unroll
  for (int r = 0; r < 8; ++r) zero8[r] = 0.0f;

  const int kend = mask ? (q0 + 16) : SEQ;     // causal: keys < q0+16 needed
  const int nblk = (kend + 31) >> 5;

  for (int blk = 0; blk < nblk; ++blk) {
    const int k32 = blk << 5;

    // speculative prefetch of next block's K/V rows (one row per lane)
    if (blk + 1 < nblk) {
      __builtin_prefetch(kb + (size_t)(k32 + 32 + lane) * FDIM, 0, 0);
      __builtin_prefetch(vb + (size_t)(k32 + 32 + lane) * FDIM, 0, 0);
    }

    // ---- K block -> two B operands (K-dim = h, padded; N = key) ----
    // B layout: elems 0..7 hold K = 8*g + 0..7; elems 8..15 zero, N = ln.
    bf16x16 B0, B1;
    cvt8(kb + (size_t)(k32 + ln) * FDIM + 8 * g,      B0, 0);
    cvt8(kb + (size_t)(k32 + 16 + ln) * FDIM + 8 * g, B1, 0);
    #pragma unroll
    for (int e = 8; e < 16; ++e) { B0[e] = (__bf16)0.0f; B1[e] = (__bf16)0.0f; }

    floatx8 s0 = wmma_bf16(Aq, B0, zero8);   // queries x keys[k32..k32+15]
    floatx8 s1 = wmma_bf16(Aq, B1, zero8);   // queries x keys[k32+16..k32+31]

    // ---- scale + causal mask + online softmax (row max via DPP) ----
    const float scale = 0.25f;               // 1/sqrt(16)
    #pragma unroll
    for (int r = 0; r < 8; ++r) {
      const int m   = r + 8 * g;             // C/D layout: row = r + 8*(lane>>4)
      const int qi  = q0 + m;
      const int k0i = k32 + ln;              // C/D layout: col = lane&15
      const int k1i = k32 + 16 + ln;
      float a0 = s0[r] * scale;
      float a1 = s1[r] * scale;
      if (mask) {
        a0 = (k0i > qi) ? -1.0e9f : a0;
        a1 = (k1i > qi) ? -1.0e9f : a1;
      }
      const float bm   = row16_max(fmaxf(a0, a1));   // block row max
      const float Mn   = fmaxf(Mrun[r], bm);
      const float corr = __expf(Mrun[r] - Mn);       // row-uniform
      const float p0   = __expf(a0 - Mn);
      const float p1   = __expf(a1 - Mn);
      Lpart[r] = Lpart[r] * corr + (p0 + p1);        // lane-local partial
      Mrun[r]  = Mn;
      O[r]    *= corr;
      lp[m * 40 + ln]      = (__bf16)p0;     // stage P for layout transpose
      lp[m * 40 + 16 + ln] = (__bf16)p1;
    }

    // lane-crossing RAW through LDS within the wave
    asm volatile("s_wait_dscnt 0" ::: "memory");

    // ---- P (16x32) in A layout: two vectorized 16B LDS reads ----
    BF16x16U up;
    up.q[0] = *(const uint4*)(lp + ln * 40 + 8 * g);        // K = 8g + 0..7
    up.q[1] = *(const uint4*)(lp + ln * 40 + 16 + 8 * g);   // K = 16 + 8g + 0..7
    const bf16x16 Ap = up.v;

    // ---- V block (32 keys x 16 h) in B layout: elem e -> K = 8*g + e (+8 if e>=8), N = ln
    bf16x16 Bv;
    #pragma unroll
    for (int e = 0; e < 16; ++e)
      Bv[e] = (__bf16)vb[(size_t)(k32 + 8 * g + e + (e >= 8 ? 8 : 0)) * FDIM + ln];

    O = wmma_bf16(Ap, Bv, O);                // accumulate P*V in fp32

    asm volatile("" ::: "memory");           // keep next block's LDS writes below reads
  }

  // ---- final row-sum reduction (once), normalize, bit-shuffle scatter ----
  #pragma unroll
  for (int r = 0; r < 8; ++r) {
    const float Lsum = row16_sum(Lpart[r]);
    const int m    = r + 8 * g;
    const int srow = q0 + m;
    const float val = O[r] / Lsum;
    const unsigned idx = (((unsigned)(j * NB + b) * SEQ + (unsigned)srow) * HD) + (unsigned)ln;
    const unsigned a  = idx >> 17;           // idx / (NB*SEQ*64)
    const unsigned b2 = (idx >> 16) & 1u;
    const unsigned s2 = (idx >> 6) & 1023u;
    const unsigned c  = idx & 63u;
    x[((size_t)b2 * SEQ + s2) * FDIM + a * 64u + c] = val;
  }
}

// ---------------------------------------------------------------------------
// Kernel 2: out = x @ W^T + bias   (rows = NB*SEQ = 2048, F = 1024)
// One wave32 per 16x16 output tile, bf16 WMMA with fp32 accumulation.
// ---------------------------------------------------------------------------
__global__ __launch_bounds__(32) void out_gemm(
    const float* __restrict__ x, const float* __restrict__ W,
    const float* __restrict__ bias, float* __restrict__ out)
{
  const int lane = threadIdx.x & 31;
  const int g    = lane >> 4;
  const int ln   = lane & 15;
  const int nt   = blockIdx.x;               // col tile 0..63
  const int mt   = blockIdx.y;               // row tile 0..127

  const float* __restrict__ xr = x + (size_t)(mt * 16 + ln) * FDIM;
  const float* __restrict__ wr = W + (size_t)(nt * 16 + ln) * FDIM;

  floatx8 acc;
  #pragma unroll
  for (int r = 0; r < 8; ++r) acc[r] = 0.0f;

  for (int k0 = 0; k0 < FDIM; k0 += 32) {
    if (k0 + 32 < FDIM) {
      __builtin_prefetch(xr + k0 + 32, 0, 0);
      __builtin_prefetch(wr + k0 + 32, 0, 0);
    }
    bf16x16 A, Bw;
    cvt8(xr + k0 + 8 * g,       A, 0);       // K = 8g + 0..7
    cvt8(xr + k0 + 16 + 8 * g,  A, 8);       // K = 16 + 8g + 0..7
    cvt8(wr + k0 + 8 * g,       Bw, 0);      // K = 8g + 0..7, N = ln
    cvt8(wr + k0 + 16 + 8 * g,  Bw, 8);      // K = 16 + 8g + 0..7
    acc = wmma_bf16(A, Bw, acc);
  }

  const float bo = bias[nt * 16 + ln];
  #pragma unroll
  for (int r = 0; r < 8; ++r) {
    const int m = r + 8 * g;
    out[(size_t)(mt * 16 + m) * FDIM + nt * 16 + ln] = acc[r] + bo;
  }
}

// ---------------------------------------------------------------------------
extern "C" void kernel_launch(void* const* d_in, const int* in_sizes, int n_in,
                              void* d_out, int out_size, void* d_ws, size_t ws_size,
                              hipStream_t stream) {
  const float* q     = (const float*)d_in[0];
  const float* k     = (const float*)d_in[1];
  const float* v     = (const float*)d_in[2];
  const float* W     = (const float*)d_in[3];
  const float* bias  = (const float*)d_in[4];
  const int*   amask = (const int*)d_in[5];
  float* out = (float*)d_out;
  float* x   = (float*)d_ws;                 // [NB, SEQ, FDIM] fp32 = 8 MB

  dim3 gridA(SEQ / 16, NB, NCH);             // 64 x 2 x 64 waves
  attn_fused<<<gridA, dim3(32), 0, stream>>>(q, k, v, amask, x);

  dim3 gridG(FDIM / 16, (NB * SEQ) / 16);    // 64 x 128 waves
  out_gemm<<<gridG, dim3(32), 0, stream>>>(x, W, bias, out);
}
